// MDFEND_9586367005058
// MI455X (gfx1250) — matrix operand order repaired and checked by reading, not compiled, not run on hardware
//
#include <hip/hip_runtime.h>
#include <hip/hip_bf16.h>

// ---------------------------------------------------------------------------
// MDFEND routed-expert network, MI455X (gfx1250, wave32, WMMA bf16).
// Strategy: counting-sort rows by domain, compute only each row's own expert.
//   x   = relu(pooled @ W_tfe + b)                         (GEMM, all rows)
//   h   = relu(xg @ W1[e] + b1)   t = relu(xg @ G1[e]+g1b) (fused, gathered)
//   df  = h  @ W2[e] + b2                                  (GEMM)
//   dr  = relu(df @ G2[e] + g2b)  -> only dotted with Gg   (GEMM, no store)
//   gl  = t.Gg1 + dr.Gg2 + ggb ; g = sigmoid(gl)
//   out = g*df + (1-g)*x   (scatter back by perm)
// A-tiles are staged to LDS with GLOBAL_LOAD_ASYNC_TO_LDS_B128 (ASYNCcnt).
// ---------------------------------------------------------------------------

typedef __bf16 bf16;
typedef __attribute__((ext_vector_type(16))) __bf16 v16bf;
typedef __attribute__((ext_vector_type(8)))  float  v8f;

static constexpr int kB  = 4096;   // batch
static constexpr int kD  = 768;    // hidden
static constexpr int kE  = 8;      // experts
static constexpr int kTM = 32;     // row tile per block
static constexpr int kAS = 776;    // LDS row stride (multiple of 8 elems)
static constexpr int kS  = 4352;   // slot space: 4096 + 8*32 padding
static constexpr int kTPE = 128;   // tiles (blocks) per expert = kB/kTM

// hdr layout (ints): [0..7]=cnt  [8..16]=off  [17..24]=fill
// ---------------------------------------------------------------------------

__device__ __forceinline__ v16bf load_frag(const bf16* p, int second_off) {
  v16bf r;
  float4* rp = reinterpret_cast<float4*>(&r);
  rp[0] = *reinterpret_cast<const float4*>(p);
  rp[1] = *reinterpret_cast<const float4*>(p + second_off);
  return r;
}

__device__ __forceinline__ v8f wmma_bf16(v16bf a, v16bf b, v8f c) {
  return __builtin_amdgcn_wmma_f32_16x16x32_bf16(
      /*neg_a=*/false, a, /*neg_b=*/false, b,
      /*c_mod=*/(short)0, c, /*reuse_a=*/false, /*reuse_b=*/false);
}

// Async copy of one 1536-byte A row into LDS, 4 threads per row, 24 x 16B
// chunks each lane. Per-lane global address supports the perm gather.
// LDS offset = low 32 bits of the generic shared-memory address (ISA 10.2).
__device__ __forceinline__ void async_fill_row(bf16* ldsRow, const bf16* gRow,
                                               int part) {
  uint32_t l0 = (uint32_t)(size_t)(void*)ldsRow;
  uint64_t g0 = (uint64_t)(size_t)(const void*)gRow;
#pragma unroll
  for (int j = 0; j < 24; ++j) {
    uint32_t c = (uint32_t)(part * 24 + j) * 16u;
    uint32_t l = l0 + c;
    uint64_t g = g0 + c;
    asm volatile("global_load_async_to_lds_b128 %0, %1, off"
                 :: "v"(l), "v"(g) : "memory");
  }
}

__device__ __forceinline__ void async_wait_all() {
  asm volatile("s_wait_asynccnt 0x0" ::: "memory");
}

// ------------------------------ setup kernels ------------------------------

__global__ void k_init(int* hdr, float* gl, int* perm) {
  int i = blockIdx.x * 256 + threadIdx.x;
  if (i < 32) hdr[i] = 0;
  if (i < kS) { gl[i] = 0.f; perm[i] = 0; }
}

__global__ void k_count(const int* __restrict__ dom, int* hdr) {
  int i = blockIdx.x * 256 + threadIdx.x;
  if (i < kB) atomicAdd(&hdr[dom[i]], 1);
}

__global__ void k_scan(int* hdr) {
  if (blockIdx.x == 0 && threadIdx.x == 0) {
    int o = 0;
    for (int e = 0; e < kE; ++e) {
      hdr[8 + e]  = o;
      hdr[17 + e] = 0;
      o += (hdr[e] + (kTM - 1)) & ~(kTM - 1);
    }
    hdr[16] = o;
  }
}

__global__ void k_fill(const int* __restrict__ dom, int* hdr, int* perm) {
  int i = blockIdx.x * 256 + threadIdx.x;
  if (i < kB) {
    int e = dom[i];
    int pos = atomicAdd(&hdr[17 + e], 1);
    perm[hdr[8 + e] + pos] = i;
  }
}

__global__ void k_cvt(const float* __restrict__ in, bf16* __restrict__ out, int n) {
  int i = blockIdx.x * 256 + threadIdx.x;
  if (i < n) out[i] = (bf16)in[i];
}

// out[e][n][k] = in[e][k][n]  (N-major, K contiguous -> B-fragment friendly)
__global__ void k_cvt_t(const float* __restrict__ in, bf16* __restrict__ out, int nmat) {
  int i = blockIdx.x * 256 + threadIdx.x;
  int total = nmat * kD * kD;
  if (i < total) {
    int e = i / (kD * kD);
    int r = i % (kD * kD);
    int n = r / kD, k = r % kD;
    out[i] = (bf16)in[e * kD * kD + k * kD + n];
  }
}

// ------------------------------ GEMM kernels -------------------------------
// Block = 128 threads (4 waves). Rows: 2 strips of 16 (strip = wave>>1),
// each strip's N range split across 2 waves (nhalf = wave&1, 24 tiles each).

__global__ void __launch_bounds__(128)
k_gemm_x(const bf16* __restrict__ Ab, const bf16* __restrict__ Wt,
         const float* __restrict__ bias, float* __restrict__ Xf,
         bf16* __restrict__ Xb) {
  __shared__ bf16 As[kTM * kAS];
  int rowBase = blockIdx.x * kTM;
  int tid = threadIdx.x;
  {
    int r = tid >> 2, part = tid & 3;
    async_fill_row(As + r * kAS, Ab + (size_t)(rowBase + r) * kD, part);
  }
  async_wait_all();
  __syncthreads();
  int lane = tid & 31, wv = tid >> 5;
  int strip = wv >> 1, nhalf = wv & 1;
  int nl = lane & 15, half = lane >> 4;
  const bf16* Arow = As + (strip * 16 + nl) * kAS + half * 16;
  for (int nt = 0; nt < 24; ++nt) {
    int n0 = nhalf * 384 + nt * 16;
    const bf16* Bp = Wt + (size_t)(n0 + nl) * kD + half * 16;
    __builtin_prefetch(Bp + 16 * kD, 0, 3);     // next n-tile of B
    v8f acc = {};
    for (int kt = 0; kt < 24; ++kt) {
      v16bf a = load_frag(Arow + kt * 32, 8);
      v16bf b = load_frag(Bp + kt * 32, 8);
      acc = wmma_bf16(a, b, acc);
    }
    int n = n0 + nl;
    float bv = bias[n];
#pragma unroll
    for (int v = 0; v < 8; ++v) {
      int m = rowBase + strip * 16 + half * 8 + v;
      float val = acc[v] + bv;
      val = val > 0.f ? val : 0.f;
      Xf[(size_t)m * kD + n] = val;
      Xb[(size_t)m * kD + n] = (bf16)val;
    }
  }
}

// Fused h + t: two WMMAs per (n,k) tile reusing the A fragment; h stored,
// t reduced against Gg[:, :D] into gl via shfl butterfly + float atomics.
__global__ void __launch_bounds__(128)
k_gemm_ht(const bf16* __restrict__ Xb, const bf16* __restrict__ W1t,
          const bf16* __restrict__ G1t, const float* __restrict__ b1,
          const float* __restrict__ g1b, const float* __restrict__ Gg,
          const int* __restrict__ perm, const int* __restrict__ hdr,
          bf16* __restrict__ Hb, float* __restrict__ gl) {
  int e = blockIdx.x >> 7, t = blockIdx.x & (kTPE - 1);
  int cnt = hdr[e];
  if (t * kTM >= cnt) return;
  int off = hdr[8 + e];
  int slotBase = off + t * kTM;
  int valid = cnt - t * kTM; if (valid > kTM) valid = kTM;

  __shared__ bf16 As[kTM * kAS];
  int tid = threadIdx.x;
  {
    int r = tid >> 2, part = tid & 3;
    int rr = (r < valid) ? r : 0;
    int grow = perm[slotBase + rr];
    async_fill_row(As + r * kAS, Xb + (size_t)grow * kD, part);
  }
  async_wait_all();
  __syncthreads();

  int lane = tid & 31, wv = tid >> 5;
  int strip = wv >> 1, nhalf = wv & 1;
  int nl = lane & 15, half = lane >> 4;
  const bf16* Arow = As + (strip * 16 + nl) * kAS + half * 16;
  const bf16* W1e = W1t + (size_t)e * kD * kD;
  const bf16* G1e = G1t + (size_t)e * kD * kD;
  const float* gg = Gg + (size_t)e * 2 * kD;   // first half: t-gate weights

  float pacc[8] = {0.f, 0.f, 0.f, 0.f, 0.f, 0.f, 0.f, 0.f};
  for (int nt = 0; nt < 24; ++nt) {
    int n0 = nhalf * 384 + nt * 16;
    const bf16* B1p = W1e + (size_t)(n0 + nl) * kD + half * 16;
    const bf16* B2p = G1e + (size_t)(n0 + nl) * kD + half * 16;
    __builtin_prefetch(B1p + 16 * kD, 0, 3);
    __builtin_prefetch(B2p + 16 * kD, 0, 3);
    v8f acc = {}, acc2 = {};
    for (int kt = 0; kt < 24; ++kt) {
      v16bf a = load_frag(Arow + kt * 32, 8);
      acc  = wmma_bf16(a, load_frag(B1p + kt * 32, 8), acc);
      acc2 = wmma_bf16(a, load_frag(B2p + kt * 32, 8), acc2);
    }
    int n = n0 + nl;
    float bh = b1[e * kD + n], bt = g1b[e * kD + n], ggn = gg[n];
#pragma unroll
    for (int v = 0; v < 8; ++v) {
      int ml = strip * 16 + half * 8 + v;
      float hv = acc[v] + bh;  hv = hv > 0.f ? hv : 0.f;
      if (ml < valid) Hb[(size_t)(slotBase + ml) * kD + n] = (bf16)hv;
      float tv = acc2[v] + bt; tv = tv > 0.f ? tv : 0.f;
      pacc[v] += tv * ggn;
    }
  }
  // reduce gate partials across the 16 lanes that share each row set
#pragma unroll
  for (int m = 1; m < 16; m <<= 1) {
#pragma unroll
    for (int v = 0; v < 8; ++v) pacc[v] += __shfl_xor(pacc[v], m, 32);
  }
  if (nl == 0) {
#pragma unroll
    for (int v = 0; v < 8; ++v) {
      int ml = strip * 16 + half * 8 + v;
      if (ml < valid) atomicAdd(&gl[slotBase + ml], pacc[v]);
    }
  }
}

__global__ void __launch_bounds__(128)
k_gemm_df(const bf16* __restrict__ Hb, const bf16* __restrict__ W2t,
          const float* __restrict__ b2, const int* __restrict__ hdr,
          float* __restrict__ DFf, bf16* __restrict__ DFb) {
  int e = blockIdx.x >> 7, t = blockIdx.x & (kTPE - 1);
  int cnt = hdr[e];
  if (t * kTM >= cnt) return;
  int off = hdr[8 + e];
  int slotBase = off + t * kTM;
  int valid = cnt - t * kTM; if (valid > kTM) valid = kTM;

  __shared__ bf16 As[kTM * kAS];
  int tid = threadIdx.x;
  {
    int r = tid >> 2, part = tid & 3;
    async_fill_row(As + r * kAS, Hb + (size_t)(slotBase + r) * kD, part);
  }
  async_wait_all();
  __syncthreads();

  int lane = tid & 31, wv = tid >> 5;
  int strip = wv >> 1, nhalf = wv & 1;
  int nl = lane & 15, half = lane >> 4;
  const bf16* Arow = As + (strip * 16 + nl) * kAS + half * 16;
  const bf16* W2e = W2t + (size_t)e * kD * kD;
  for (int nt = 0; nt < 24; ++nt) {
    int n0 = nhalf * 384 + nt * 16;
    const bf16* Bp = W2e + (size_t)(n0 + nl) * kD + half * 16;
    __builtin_prefetch(Bp + 16 * kD, 0, 3);
    v8f acc = {};
    for (int kt = 0; kt < 24; ++kt) {
      v16bf a = load_frag(Arow + kt * 32, 8);
      acc = wmma_bf16(a, load_frag(Bp + kt * 32, 8), acc);
    }
    int n = n0 + nl;
    float bv = b2[e * kD + n];
#pragma unroll
    for (int v = 0; v < 8; ++v) {
      int ml = strip * 16 + half * 8 + v;
      if (ml < valid) {
        float val = acc[v] + bv;                   // no relu on df
        DFf[(size_t)(slotBase + ml) * kD + n] = val;
        DFb[(size_t)(slotBase + ml) * kD + n] = (bf16)val;
      }
    }
  }
}

// dr is never materialized: relu(df@G2+g2b) dotted with Gg[:, D:] -> gl.
__global__ void __launch_bounds__(128)
k_gemm_dr(const bf16* __restrict__ DFb, const bf16* __restrict__ G2t,
          const float* __restrict__ g2b, const float* __restrict__ Gg,
          const int* __restrict__ hdr, float* __restrict__ gl) {
  int e = blockIdx.x >> 7, t = blockIdx.x & (kTPE - 1);
  int cnt = hdr[e];
  if (t * kTM >= cnt) return;
  int off = hdr[8 + e];
  int slotBase = off + t * kTM;
  int valid = cnt - t * kTM; if (valid > kTM) valid = kTM;

  __shared__ bf16 As[kTM * kAS];
  int tid = threadIdx.x;
  {
    int r = tid >> 2, part = tid & 3;
    async_fill_row(As + r * kAS, DFb + (size_t)(slotBase + r) * kD, part);
  }
  async_wait_all();
  __syncthreads();

  int lane = tid & 31, wv = tid >> 5;
  int strip = wv >> 1, nhalf = wv & 1;
  int nl = lane & 15, half = lane >> 4;
  const bf16* Arow = As + (strip * 16 + nl) * kAS + half * 16;
  const bf16* G2e = G2t + (size_t)e * kD * kD;
  const float* gg = Gg + (size_t)e * 2 * kD + kD;   // second half: dr-gate weights

  float pacc[8] = {0.f, 0.f, 0.f, 0.f, 0.f, 0.f, 0.f, 0.f};
  for (int nt = 0; nt < 24; ++nt) {
    int n0 = nhalf * 384 + nt * 16;
    const bf16* Bp = G2e + (size_t)(n0 + nl) * kD + half * 16;
    __builtin_prefetch(Bp + 16 * kD, 0, 3);
    v8f acc = {};
    for (int kt = 0; kt < 24; ++kt) {
      v16bf a = load_frag(Arow + kt * 32, 8);
      acc = wmma_bf16(a, load_frag(Bp + kt * 32, 8), acc);
    }
    int n = n0 + nl;
    float bv = g2b[e * kD + n], ggn = gg[n];
#pragma unroll
    for (int v = 0; v < 8; ++v) {
      float dv = acc[v] + bv; dv = dv > 0.f ? dv : 0.f;
      pacc[v] += dv * ggn;
    }
  }
#pragma unroll
  for (int m = 1; m < 16; m <<= 1) {
#pragma unroll
    for (int v = 0; v < 8; ++v) pacc[v] += __shfl_xor(pacc[v], m, 32);
  }
  if (nl == 0) {
#pragma unroll
    for (int v = 0; v < 8; ++v) {
      int ml = strip * 16 + half * 8 + v;
      if (ml < valid) atomicAdd(&gl[slotBase + ml], pacc[v]);
    }
  }
}

__global__ void k_final(const float* __restrict__ DFf, const float* __restrict__ gl,
                        const float* __restrict__ ggb, const int* __restrict__ perm,
                        const int* __restrict__ hdr, const float* __restrict__ Xf,
                        float* __restrict__ out) {
  int idx = blockIdx.x * 256 + threadIdx.x;
  if (idx >= kS * kD) return;
  int slot = idx / kD, d = idx % kD;
  int e = -1;
  for (int i = 0; i < kE; ++i) {
    int o = hdr[8 + i], c = hdr[i];
    if (slot >= o && slot < o + c) { e = i; break; }
  }
  if (e < 0) return;                               // padding slot
  int row = perm[slot];
  float g = 1.f / (1.f + __expf(-(gl[slot] + ggb[e])));
  out[(size_t)row * kD + d] = g * DFf[idx] + (1.f - g) * Xf[(size_t)row * kD + d];
}

// ------------------------------- launcher ----------------------------------

extern "C" void kernel_launch(void* const* d_in, const int* in_sizes, int n_in,
                              void* d_out, int out_size, void* d_ws, size_t ws_size,
                              hipStream_t stream) {
  (void)in_sizes; (void)n_in; (void)out_size; (void)ws_size;
  const float* pooled = (const float*)d_in[0];
  const int*   dom    = (const int*)d_in[1];
  const float* W_tfe  = (const float*)d_in[2];
  const float* b_tfe  = (const float*)d_in[3];
  const float* W1     = (const float*)d_in[4];
  const float* b1     = (const float*)d_in[5];
  const float* W2     = (const float*)d_in[6];
  const float* b2     = (const float*)d_in[7];
  const float* G1     = (const float*)d_in[8];
  const float* g1b    = (const float*)d_in[9];
  const float* G2     = (const float*)d_in[10];
  const float* g2b    = (const float*)d_in[11];
  const float* Gg     = (const float*)d_in[12];
  const float* ggb    = (const float*)d_in[13];
  float* out = (float*)d_out;

  char* w = (char*)d_ws;
  auto alloc = [&](size_t bytes) -> char* {
    char* p = w; w += (bytes + 255) & ~(size_t)255; return p;
  };
  const size_t DD = (size_t)kD * kD;
  int*   hdr     = (int*)  alloc(128);
  int*   perm    = (int*)  alloc((size_t)kS * 4);
  float* gl      = (float*)alloc((size_t)kS * 4);
  bf16*  pooledb = (bf16*) alloc((size_t)kB * kD * 2);
  bf16*  Wtfet   = (bf16*) alloc(DD * 2);
  bf16*  W1t     = (bf16*) alloc((size_t)kE * DD * 2);
  bf16*  W2t     = (bf16*) alloc((size_t)kE * DD * 2);
  bf16*  G1t     = (bf16*) alloc((size_t)kE * DD * 2);
  bf16*  G2t     = (bf16*) alloc((size_t)kE * DD * 2);
  float* Xf      = (float*)alloc((size_t)kB * kD * 4);
  bf16*  Xb      = (bf16*) alloc((size_t)kB * kD * 2);
  bf16*  Hb      = (bf16*) alloc((size_t)kS * kD * 2);
  float* DFf     = (float*)alloc((size_t)kS * kD * 4);
  bf16*  DFb     = (bf16*) alloc((size_t)kS * kD * 2);

  // routing
  k_init <<<(kS + 255) / 256, 256, 0, stream>>>(hdr, gl, perm);
  k_count<<<kB / 256, 256, 0, stream>>>(dom, hdr);
  k_scan <<<1, 32, 0, stream>>>(hdr);
  k_fill <<<kB / 256, 256, 0, stream>>>(dom, hdr, perm);

  // precision / layout conversion (fp32 -> bf16, weights transposed N-major)
  k_cvt  <<<(kB * kD) / 256, 256, 0, stream>>>(pooled, pooledb, kB * kD);
  k_cvt_t<<<(int)(DD / 256), 256, 0, stream>>>(W_tfe, Wtfet, 1);
  k_cvt_t<<<(int)(kE * DD / 256), 256, 0, stream>>>(W1, W1t, kE);
  k_cvt_t<<<(int)(kE * DD / 256), 256, 0, stream>>>(W2, W2t, kE);
  k_cvt_t<<<(int)(kE * DD / 256), 256, 0, stream>>>(G1, G1t, kE);
  k_cvt_t<<<(int)(kE * DD / 256), 256, 0, stream>>>(G2, G2t, kE);

  // WMMA pipeline
  k_gemm_x <<<kB / kTM, 128, 0, stream>>>(pooledb, Wtfet, b_tfe, Xf, Xb);
  k_gemm_ht<<<kE * kTPE, 128, 0, stream>>>(Xb, W1t, G1t, b1, g1b, Gg, perm, hdr, Hb, gl);
  k_gemm_df<<<kE * kTPE, 128, 0, stream>>>(Hb, W2t, b2, hdr, DFf, DFb);
  k_gemm_dr<<<kE * kTPE, 128, 0, stream>>>(DFb, G2t, g2b, Gg, hdr, gl);

  // gate + blend + scatter
  k_final<<<(kS * kD) / 256, 256, 0, stream>>>(DFf, gl, ggb, perm, hdr, Xf, out);
}
